// Generator_38543036514875
// MI455X (gfx1250) — matrix operand order repaired and checked by reading, not compiled, not run on hardware
//
#include <hip/hip_runtime.h>
#include <cstdint>
#include <cstddef>

typedef __attribute__((ext_vector_type(16))) __bf16 bf16x16;
typedef __attribute__((ext_vector_type(8)))  __bf16 bf16x8;
typedef __attribute__((ext_vector_type(8)))  float  f32x8;

#define B_SZ 16
#define WDIM 512

__device__ __forceinline__ float lrelu02(float x){ return x > 0.f ? x : 0.2f*x; }

// ---------------- mapping network ----------------
__global__ void k_fc_lrelu(const float* __restrict__ x, const float* __restrict__ W,
                           const float* __restrict__ bias, float* __restrict__ y, int K)
{
  int idx = blockIdx.x*blockDim.x + threadIdx.x;
  if (idx >= B_SZ*WDIM) return;
  int b = idx / WDIM, o = idx % WDIM;
  const float* xr = x + (size_t)b*K;
  const float* wr = W + (size_t)o*K;
  float acc = 0.f;
  for (int k=0;k<K;k++) acc += xr[k]*wr[k];
  y[idx] = lrelu02(acc + bias[o]);
}

__global__ void k_pixel_norm(const float* __restrict__ x, float* __restrict__ y)
{
  __shared__ float red[256];
  int b = blockIdx.x, t = threadIdx.x;
  float v0 = x[b*WDIM + t], v1 = x[b*WDIM + 256 + t];
  red[t] = v0*v0 + v1*v1;
  __syncthreads();
  for (int s=128; s>0; s>>=1){ if (t<s) red[t]+=red[t+s]; __syncthreads(); }
  float inv = rsqrtf(red[0]*(1.0f/WDIM) + 1e-8f);
  y[b*WDIM+t] = v0*inv; y[b*WDIM+256+t] = v1*inv;
}

// ---------------- modulation scalars ----------------
__global__ void k_mod_scale(const float* __restrict__ wlat, const float* __restrict__ mw,
                            const float* __restrict__ mb, float* __restrict__ s, int I)
{
  int idx = blockIdx.x*blockDim.x + threadIdx.x;
  if (idx >= B_SZ*I) return;
  int b = idx / I, i = idx % I;
  const float* wr = wlat + (size_t)b*WDIM;
  const float* mr = mw   + (size_t)i*WDIM;
  float acc = 0.f;
  for (int k=0;k<WDIM;k++) acc += wr[k]*mr[k];
  s[idx] = acc + mb[i] + 1.0f;
}

__global__ void k_weight_sq(const float* __restrict__ W, float* __restrict__ wsq, int I, int O)
{
  int idx = blockIdx.x*blockDim.x + threadIdx.x;
  if (idx >= O*I) return;
  const float* wr = W + (size_t)idx*9;
  float acc = 0.f;
  for (int t=0;t<9;t++) acc += wr[t]*wr[t];
  wsq[idx] = acc;
}

__global__ void k_demod(const float* __restrict__ s, const float* __restrict__ wsq,
                        float* __restrict__ d, int I, int O)
{
  int idx = blockIdx.x*blockDim.x + threadIdx.x;
  if (idx >= B_SZ*O) return;
  int b = idx / O, o = idx % O;
  const float* sr = s   + (size_t)b*I;
  const float* wr = wsq + (size_t)o*I;
  float acc = 0.f;
  for (int i=0;i<I;i++){ float sv = sr[i]; acc += sv*sv*wr[i]; }
  d[idx] = rsqrtf(acc + 1e-8f);
}

// weights (O,I,3,3) f32 -> (9,O,I) bf16 (tap-major, K=I contiguous)
__global__ void k_cvt_w(const float* __restrict__ src, __bf16* __restrict__ dst,
                        int lgI, int lgO)
{
  int idx = blockIdx.x*blockDim.x + threadIdx.x;        // enumerates dst
  int I = 1 << lgI, O = 1 << lgO;
  if (idx >= (O<<lgI)*9) return;
  int i  = idx & (I-1);
  int t2 = idx >> lgI;
  int o  = t2 & (O-1);
  int t  = t2 >> lgO;
  dst[idx] = (__bf16)src[(((size_t)o<<lgI) + i)*9 + t];
}

__global__ void k_zero_bf16(__bf16* __restrict__ dst, int n)
{
  int idx = blockIdx.x*blockDim.x + threadIdx.x;
  if (idx < n) dst[idx] = (__bf16)0.0f;
}

// dst (B,r,r,I) NHWC bf16 = src(NCHW f32, res r>>sh, batch stride srcB) * s[b,i]
__global__ void k_modulate_nhwc(const float* __restrict__ src, const float* __restrict__ s,
                                __bf16* __restrict__ dst, int lgI, int lgR, int sh, int srcB)
{
  int idx = blockIdx.x*blockDim.x + threadIdx.x;
  int I = 1 << lgI, r = 1 << lgR, rr = 1 << (2*lgR);
  int total = (B_SZ << (2*lgR)) << lgI;
  if (idx >= total) return;
  int i = idx & (I-1);
  int t = idx >> lgI;
  int p = t & (rr-1);
  int b = t >> (2*lgR);
  int h = p >> lgR, w = p & (r-1);
  int rs = r >> sh;
  size_t si = (size_t)b*srcB + (((size_t)i*rs + (h>>sh))*rs) + (w>>sh);
  dst[idx] = (__bf16)(src[si] * s[(b<<lgI) + i]);
}

// -------- 3x3 conv: 9 shifted 1x1 GEMMs, WMMA bf16, 2x2 register blocking ----
// xs: (B,r,r,I) NHWC bf16, wq: (9,O,I) bf16, zpad: >=I zeros
// Each wave computes a 32(M=O) x 32(N=spatial) tile: 4 accumulators.
// grid: (O/32, r*r/64, B), block 64 (2 waves; wave -> n-tile pair)
__global__ void __launch_bounds__(64)
k_conv3x3_wmma(const __bf16* __restrict__ xs,
               const __bf16* __restrict__ wq,
               const __bf16* __restrict__ zpad,
               const float* __restrict__ dmod,
               const float* __restrict__ ng,
               const float* __restrict__ noise,
               float* __restrict__ out,
               int lgR, int I, int O)
{
  const int lane   = threadIdx.x & 31;
  const int wv     = threadIdx.x >> 5;
  const int mt     = blockIdx.x;              // 32-row M tile
  const int nt     = blockIdx.y * 2 + wv;     // 32-col N tile
  const int b      = blockIdx.z;
  const int r      = 1 << lgR;
  const int rr     = 1 << (2*lgR);
  const int lane16 = lane & 15;
  const int hi     = lane >> 4;               // 0 or 1

  const int p0 = nt*32 + lane16;              // two spatial columns per lane
  const int p1 = p0 + 16;
  const int h0 = p0 >> lgR, w0 = p0 & (r-1);
  const int h1 = p1 >> lgR, w1 = p1 & (r-1);

  const __bf16* xb  = xs + ((size_t)b*rr)*I;
  const int o_a = mt*32 + lane16;             // A rows: o_a and o_a+16

  f32x8 c00 = {}, c01 = {}, c10 = {}, c11 = {};

  #pragma unroll
  for (int tap = 0; tap < 9; ++tap){
    const int kh = tap/3, kw = tap - (tap/3)*3;   // constant-folded (unrolled)
    const int hh0 = h0 + kh - 1, ww0 = w0 + kw - 1;
    const int hh1 = h1 + kh - 1, ww1 = w1 + kw - 1;
    const bool v0 = ((unsigned)hh0 < (unsigned)r) & ((unsigned)ww0 < (unsigned)r);
    const bool v1 = ((unsigned)hh1 < (unsigned)r) & ((unsigned)ww1 < (unsigned)r);
    const __bf16* brow0 = v0 ? (xb + (ptrdiff_t)(hh0*r + ww0)*I) : zpad;
    const __bf16* brow1 = v1 ? (xb + (ptrdiff_t)(hh1*r + ww1)*I) : zpad;
    const __bf16* arow0 = wq + ((size_t)tap*O + o_a)*I;
    const __bf16* arow1 = arow0 + (size_t)16*I;

    #pragma unroll 2
    for (int k0 = 0; k0 < I; k0 += 32){
      // A fragments: 16-bit A 16x32 layout (lanes 0-15: K 0-7/16-23; 16-31: K 8-15/24-31)
      const int ka = k0 + hi*8;
      bf16x8 a0l = *(const bf16x8*)(arow0 + ka);
      bf16x8 a0h = *(const bf16x8*)(arow0 + ka + 16);
      bf16x8 a1l = *(const bf16x8*)(arow1 + ka);
      bf16x8 a1h = *(const bf16x8*)(arow1 + ka + 16);
      // B fragments: 16-bit B 32x16 layout (lanes 0-15: K 0-7/16-23; 16-31: K 8-15/24-31)
      const int kb = k0 + hi*8;
      bf16x8 b0l = *(const bf16x8*)(brow0 + kb);
      bf16x8 b0h = *(const bf16x8*)(brow0 + kb + 16);
      bf16x8 b1l = *(const bf16x8*)(brow1 + kb);
      bf16x8 b1h = *(const bf16x8*)(brow1 + kb + 16);

      bf16x16 a0, a1, b0, b1;
      #pragma unroll
      for (int j=0;j<8;j++){
        a0[j] = a0l[j]; a0[8+j] = a0h[j];
        a1[j] = a1l[j]; a1[8+j] = a1h[j];
        b0[j] = b0l[j]; b0[8+j] = b0h[j];
        b1[j] = b1l[j]; b1[8+j] = b1h[j];
      }
      c00 = __builtin_amdgcn_wmma_f32_16x16x32_bf16(false, a0, false, b0, (short)0, c00, false, false);
      c01 = __builtin_amdgcn_wmma_f32_16x16x32_bf16(false, a0, false, b1, (short)0, c01, false, false);
      c10 = __builtin_amdgcn_wmma_f32_16x16x32_bf16(false, a1, false, b0, (short)0, c10, false, false);
      c11 = __builtin_amdgcn_wmma_f32_16x16x32_bf16(false, a1, false, b1, (short)0, c11, false, false);
    }
  }

  // C/D layout: VGPR v holds (M = v + 8*hi, N = lane&15)
  const float* drow = dmod + (size_t)b*O;
  #pragma unroll
  for (int mi=0; mi<2; mi++){
    #pragma unroll
    for (int ni=0; ni<2; ni++){
      f32x8 cv = mi ? (ni ? c11 : c10) : (ni ? c01 : c00);
      const int pp = nt*32 + ni*16 + lane16;
      #pragma unroll
      for (int v=0; v<8; v++){
        int o = mt*32 + mi*16 + hi*8 + v;
        size_t oi = ((size_t)b*O + o)*rr + pp;
        float y = cv[v]*drow[o] + ng[o]*noise[oi];
        out[oi] = lrelu02(y);
      }
    }
  }
}

// ---------------- toRGB (1x1 modulated conv, no demod) + up2 accumulate ------
__global__ void k_rgb_acc(const float* __restrict__ act, const float* __restrict__ srgb,
                          const float* __restrict__ wrgb, const float* __restrict__ prev,
                          float* __restrict__ outp, int O, int lgR, int hasPrev)
{
  int idx = blockIdx.x*blockDim.x + threadIdx.x;
  int r = 1 << lgR, rr = 1 << (2*lgR);
  if (idx >= (B_SZ<<(2*lgR))) return;
  int b = idx >> (2*lgR);
  int p = idx & (rr-1);
  int h = p >> lgR, w = p & (r-1);
  const float* ab = act  + ((size_t)b*O)*rr + p;
  const float* sb = srgb + (size_t)b*O;
  float acc = 0.f;
  for (int o=0;o<O;o++) acc += ab[(size_t)o*rr] * sb[o] * wrgb[o];
  if (hasPrev){
    int rp = r >> 1;
    acc += prev[(size_t)b*rp*rp + (h>>1)*rp + (w>>1)];
  }
  outp[idx] = acc;
}

__global__ void k_tanh(const float* __restrict__ x, float* __restrict__ y, int n)
{
  int idx = blockIdx.x*blockDim.x + threadIdx.x;
  if (idx < n) y[idx] = tanhf(x[idx]);
}

static inline int ilog2(int v){ int l=0; while ((1<<l) < v) ++l; return l; }

// ---------------- host orchestration ----------------
extern "C" void kernel_launch(void* const* d_in, const int* in_sizes, int n_in,
                              void* d_out, int out_size, void* d_ws, size_t ws_size,
                              hipStream_t stream)
{
  (void)in_sizes; (void)n_in; (void)out_size; (void)ws_size;
  const int CH[5] = {1024,512,256,128,64};

  const float* mapW[8]; const float* mapBias[8];
  for (int l=0;l<8;l++){ mapW[l]=(const float*)d_in[2*l]; mapBias[l]=(const float*)d_in[2*l+1]; }
  const float* constC = (const float*)d_in[16];
  const float* z      = (const float*)d_in[72];

  char* ws = (char*)d_ws;
  size_t off = 0;
  auto ALLOC = [&](size_t bytes)->char*{
    size_t a = (off + 255) & ~(size_t)255;
    off = a + bytes;
    return ws + a;
  };
  float*  MAP0 = (float*)ALLOC((size_t)B_SZ*WDIM*4);
  float*  MAP1 = (float*)ALLOC((size_t)B_SZ*WDIM*4);
  float*  WLAT = (float*)ALLOC((size_t)B_SZ*WDIM*4);
  float*  S1   = (float*)ALLOC((size_t)B_SZ*1024*4);
  float*  S2   = (float*)ALLOC((size_t)B_SZ*1024*4);
  float*  SRGB = (float*)ALLOC((size_t)B_SZ*1024*4);
  float*  DMOD = (float*)ALLOC((size_t)B_SZ*1024*4);
  __bf16* ZPAD = (__bf16*)ALLOC((size_t)2048*2);
  float*  WSQ  = (float*)ALLOC((size_t)1024*1024*4);
  __bf16* WQ   = (__bf16*)ALLOC((size_t)1024*1024*9*2);
  float*  OUTA = (float*)ALLOC((size_t)B_SZ*128*128*4);
  float*  OUTB = (float*)ALLOC((size_t)B_SZ*128*128*4);
  float*  ACTP = (float*)ALLOC((size_t)B_SZ*64*128*128*4);   // 67 MB
  float*  ACTT = (float*)ALLOC((size_t)B_SZ*64*128*128*4);   // 67 MB
  __bf16* XS   = (__bf16*)ALLOC((size_t)B_SZ*128*128*128*2); // 67 MB

  k_zero_bf16<<<dim3(8), dim3(256), 0, stream>>>(ZPAD, 2048);

  // ---- mapping network ----
  {
    dim3 blk(256), grd((B_SZ*WDIM+255)/256);
    const float* cur = z;
    float* buf[2] = {MAP0, MAP1};
    int K = 128;
    for (int l=0;l<8;l++){
      float* dst = buf[l&1];
      k_fc_lrelu<<<grd, blk, 0, stream>>>(cur, mapW[l], mapBias[l], dst, K);
      cur = dst; K = WDIM;
    }
    k_pixel_norm<<<dim3(B_SZ), dim3(256), 0, stream>>>(cur, WLAT);
  }

  // ---- synthesis blocks ----
  int inC = 1024, r = 4;
  const float* prevOut = nullptr;
  float* outPing = OUTA; float* outPong = OUTB;
  for (int l=0;l<5;l++){
    int O = CH[l]; r *= 2; int rr = r*r;
    int lgR = ilog2(r), lgI = ilog2(inC), lgO = ilog2(O);
    int bi = 17 + 8*l, ri = 57 + 3*l, ni = 73 + 2*l;
    const float* c1w  = (const float*)d_in[bi+0];
    const float* c1mw = (const float*)d_in[bi+1];
    const float* c1mb = (const float*)d_in[bi+2];
    const float* c2w  = (const float*)d_in[bi+3];
    const float* c2mw = (const float*)d_in[bi+4];
    const float* c2mb = (const float*)d_in[bi+5];
    const float* ng1  = (const float*)d_in[bi+6];
    const float* ng2  = (const float*)d_in[bi+7];
    const float* rgbw = (const float*)d_in[ri+0];
    const float* rgbmw= (const float*)d_in[ri+1];
    const float* rgbmb= (const float*)d_in[ri+2];
    const float* n1   = (const float*)d_in[ni+0];
    const float* n2   = (const float*)d_in[ni+1];

    // conv1: upsample+modulate (NHWC bf16) -> WMMA conv -> demod+noise+lrelu
    k_mod_scale<<<dim3((B_SZ*inC+255)/256), dim3(256), 0, stream>>>(WLAT, c1mw, c1mb, S1, inC);
    k_weight_sq<<<dim3((O*inC+255)/256), dim3(256), 0, stream>>>(c1w, WSQ, inC, O);
    k_demod<<<dim3((B_SZ*O+255)/256), dim3(256), 0, stream>>>(S1, WSQ, DMOD, inC, O);
    k_cvt_w<<<dim3((O*inC*9+255)/256), dim3(256), 0, stream>>>(c1w, WQ, lgI, lgO);
    {
      int total = B_SZ*inC*rr;
      const float* src = (l==0) ? constC : ACTP;
      int srcB = (l==0) ? 0 : inC*(r/2)*(r/2);
      k_modulate_nhwc<<<dim3((total+255)/256), dim3(256), 0, stream>>>(src, S1, XS, lgI, lgR, 1, srcB);
    }
    k_conv3x3_wmma<<<dim3(O/32, rr/64, B_SZ), dim3(64), 0, stream>>>(
        XS, WQ, ZPAD, DMOD, ng1, n1, ACTT, lgR, inC, O);

    // conv2
    k_mod_scale<<<dim3((B_SZ*O+255)/256), dim3(256), 0, stream>>>(WLAT, c2mw, c2mb, S2, O);
    k_weight_sq<<<dim3((O*O+255)/256), dim3(256), 0, stream>>>(c2w, WSQ, O, O);
    k_demod<<<dim3((B_SZ*O+255)/256), dim3(256), 0, stream>>>(S2, WSQ, DMOD, O, O);
    k_cvt_w<<<dim3((O*O*9+255)/256), dim3(256), 0, stream>>>(c2w, WQ, lgO, lgO);
    {
      int total = B_SZ*O*rr;
      k_modulate_nhwc<<<dim3((total+255)/256), dim3(256), 0, stream>>>(ACTT, S2, XS, lgO, lgR, 0, O*rr);
    }
    k_conv3x3_wmma<<<dim3(O/32, rr/64, B_SZ), dim3(64), 0, stream>>>(
        XS, WQ, ZPAD, DMOD, ng2, n2, ACTP, lgR, O, O);

    // toRGB + skip accumulate
    k_mod_scale<<<dim3((B_SZ*O+255)/256), dim3(256), 0, stream>>>(WLAT, rgbmw, rgbmb, SRGB, O);
    k_rgb_acc<<<dim3((B_SZ*rr+255)/256), dim3(256), 0, stream>>>(
        ACTP, SRGB, rgbw, prevOut, outPing, O, lgR, (l>0) ? 1 : 0);
    prevOut = outPing;
    { float* t = outPing; outPing = outPong; outPong = t; }
    inC = O;
  }

  k_tanh<<<dim3((B_SZ*128*128+255)/256), dim3(256), 0, stream>>>(
      prevOut, (float*)d_out, B_SZ*128*128);
}
